// MultiHeadSelfAttention_24713241821224
// MI455X (gfx1250) — matrix operand order repaired and checked by reading, not compiled, not run on hardware
//
#include <hip/hip_runtime.h>
#include <hip/hip_bf16.h>

typedef __attribute__((ext_vector_type(16))) __bf16 bf16x16;
typedef __attribute__((ext_vector_type(8)))  __bf16 bf16x8;
typedef __attribute__((ext_vector_type(8)))  float  f32x8;
typedef unsigned int u32x4 __attribute__((ext_vector_type(4)));
typedef int          i32x8 __attribute__((ext_vector_type(8)));
typedef int          i32x4 __attribute__((ext_vector_type(4)));

#define D_MODEL   1024
#define NUM_HEADS 16
#define D_HEAD    64
#define QKV_LD    3072   // fused Q|K|V row stride (elements)

// ---------------------------------------------------------------------------
// WMMA helpers (CDNA5 16x16x32 bf16, fp32 accumulate)
// ---------------------------------------------------------------------------
static __device__ __forceinline__ f32x8 wmma_bf16(bf16x16 a, bf16x16 b, f32x8 c) {
    // (neg_a, A, neg_b, B, c_mod, C, reuse_a, reuse_b)
    return __builtin_amdgcn_wmma_f32_16x16x32_bf16(false, a, false, b, (short)0, c, false, false);
}

// A-fragment 16x32 (MxK), row-major source with leading dim lda (elements).
// Lane l: row = l&15 ; holds k = (l>>4)*16 + {0..15} (two 8-elem chunks).
static __device__ __forceinline__ bf16x16 load_a_frag(const __bf16* base, int lda, int lane) {
    int m    = lane & 15;
    int half = lane >> 4;
    const __bf16* p = base + (size_t)m * lda + half * 16;
    bf16x8 lo = *(const bf16x8*)(p);
    bf16x8 hi = *(const bf16x8*)(p + 8);
    return __builtin_shufflevector(lo, hi, 0,1,2,3,4,5,6,7,8,9,10,11,12,13,14,15);
}

// B-fragment 32x16 (KxN) for NT GEMM: B[k,n] = Brow[n][k], Brow row-major, ld ldb.
// Lane l: col n = l&15 ; holds k = (l>>4)*16 + {0..15} -> contiguous 32B.
static __device__ __forceinline__ bf16x16 load_b_frag(const __bf16* base, int ldb, int lane) {
    int n  = lane & 15;
    int k0 = (lane >> 4) * 16;
    return *(const bf16x16*)(base + (size_t)n * ldb + k0);
}

// ---------------------------------------------------------------------------
// Tensor Data Mover: DMA a 2-D tile (tile_rows x 32 bf16, row stride K elems)
// from global memory into LDS at byte offset lds_off. D# encoding per CDNA5
// ISA 8.3/8.4: group0 = {flags, lds_addr, global_addr, type=2},
// group1 = {data_size=2B, tensor dims, tile dims, dim0 stride}.
// ---------------------------------------------------------------------------
static __device__ __forceinline__ void tdm_load_2d(unsigned lds_off,
                                                   const void* gptr,
                                                   unsigned k_elems,
                                                   unsigned tile_rows) {
    unsigned long long ga = (unsigned long long)(size_t)gptr;
    u32x4 g0;
    g0[0] = 1u;                                          // count=1, not restore
    g0[1] = lds_off;                                     // lds_addr
    g0[2] = (unsigned)(ga & 0xFFFFFFFFu);                // global_addr[31:0]
    g0[3] = (unsigned)((ga >> 32) & 0x1FFFFFFu) | (2u << 30); // addr[56:32] | type=2

    unsigned T0 = k_elems;        // tensor_dim0 (K, contiguous)
    unsigned T1 = 1u << 20;       // tensor_dim1 (large; no OOB expected)
    unsigned S0 = k_elems;        // tensor_dim0_stride = row stride
    i32x8 g1;
    g1[0] = (int)(1u << 16);                              // data_size = 2 bytes
    g1[1] = (int)((T0 & 0xFFFFu) << 16);                  // tensor_dim0[15:0]
    g1[2] = (int)((T0 >> 16) | ((T1 & 0xFFFFu) << 16));   // dim0[31:16] | dim1[15:0]
    g1[3] = (int)((T1 >> 16) | (32u << 16));              // dim1[31:16] | tile_dim0=32
    g1[4] = (int)(tile_rows & 0xFFFFu);                   // tile_dim1 ; tile_dim2=0
    g1[5] = (int)S0;                                      // dim0_stride[31:0]
    g1[6] = 0;                                            // dim0_stride[47:32] | dim1_stride lo
    g1[7] = 0;
    i32x4 z4 = {0, 0, 0, 0};
#if defined(__clang_major__) && (__clang_major__ >= 23)
    i32x8 z8 = {0, 0, 0, 0, 0, 0, 0, 0};
    __builtin_amdgcn_tensor_load_to_lds(g0, g1, z4, z4, z8, 0);
#else
    __builtin_amdgcn_tensor_load_to_lds(g0, g1, z4, z4, 0);
#endif
}

// ---------------------------------------------------------------------------
// fp32 -> bf16 conversion
// ---------------------------------------------------------------------------
__global__ __launch_bounds__(256)
void cvt_f32_bf16(const float* __restrict__ s, __bf16* __restrict__ d, int n) {
    int i = blockIdx.x * 256 + threadIdx.x;
    if (i < n) d[i] = (__bf16)s[i];
}

// ---------------------------------------------------------------------------
// NT GEMM: C[M,N] = A[M,K] * Brow[N,K]^T.
// 256 threads = 8 waves; block tile 128x64 (waves 4x2, each 32x32).
// K-tiles are staged into LDS by the Tensor Data Mover (wave 0 issues the
// descriptors, waits on TENSORcnt, then the block barrier releases compute).
// ---------------------------------------------------------------------------
template<bool STORE_BF16>
__global__ __launch_bounds__(256)
void gemm_nt(const __bf16* __restrict__ A, const __bf16* __restrict__ B,
             void* __restrict__ Cout, int M, int N, int K) {
    __shared__ __bf16 tileA[128][32];   // 8 KB
    __shared__ __bf16 tileB[64][32];    // 4 KB

    int lane = threadIdx.x & 31;
    int w    = threadIdx.x >> 5;
    int wm   = w >> 1, wn = w & 1;
    int m0   = blockIdx.y * 128;
    int n0   = blockIdx.x * 64;

    unsigned offA = (unsigned)(size_t)&tileA[0][0];   // LDS aperture: addr[31:0]
    unsigned offB = (unsigned)(size_t)&tileB[0][0];

    f32x8 acc[2][2] = {};

    for (int kk = 0; kk < K; kk += 32) {
        if (w == 0) {
            tdm_load_2d(offA, A + (size_t)m0 * K + kk, (unsigned)K, 128u);
            tdm_load_2d(offB, B + (size_t)n0 * K + kk, (unsigned)K, 64u);
            __builtin_amdgcn_s_wait_tensorcnt(0);
        }
        __syncthreads();

        bf16x16 a0 = load_a_frag(&tileA[wm * 32][0],      32, lane);
        bf16x16 a1 = load_a_frag(&tileA[wm * 32 + 16][0], 32, lane);
        bf16x16 b0 = load_b_frag(&tileB[wn * 32][0],      32, lane);
        bf16x16 b1 = load_b_frag(&tileB[wn * 32 + 16][0], 32, lane);
        acc[0][0] = wmma_bf16(a0, b0, acc[0][0]);
        acc[0][1] = wmma_bf16(a0, b1, acc[0][1]);
        acc[1][0] = wmma_bf16(a1, b0, acc[1][0]);
        acc[1][1] = wmma_bf16(a1, b1, acc[1][1]);
        __syncthreads();
    }

    int half = lane >> 4, nn = lane & 15;
#pragma unroll
    for (int mi = 0; mi < 2; ++mi)
#pragma unroll
        for (int ni = 0; ni < 2; ++ni)
#pragma unroll
            for (int r = 0; r < 8; ++r) {
                size_t row = (size_t)(m0 + wm * 32 + mi * 16 + r + 8 * half);
                size_t col = (size_t)(n0 + wn * 32 + ni * 16 + nn);
                if (STORE_BF16)
                    ((__bf16*)Cout)[row * N + col] = (__bf16)acc[mi][ni][r];
                else
                    ((float*)Cout)[row * N + col] = acc[mi][ni][r];
            }
}

// ---------------------------------------------------------------------------
// RoPE on Q and K columns of the fused QKV buffer (interleaved pairs).
// ---------------------------------------------------------------------------
__global__ __launch_bounds__(256)
void rope_qk(__bf16* __restrict__ qkv, const int* __restrict__ pos, int S) {
    int t = blockIdx.x * 256 + threadIdx.x;
    int p = t & 31; t >>= 5;            // pair index 0..31
    int h = t & 15; t >>= 4;            // head
    int s = t % S;  int b = t / S;
    float inv = __expf(-0.28782313662425572f * (float)p);  // 10000^(-p/32)
    float ang = (float)pos[s] * inv;
    float c = __cosf(ang), sn = __sinf(ang);
    size_t base = ((size_t)(b * S + s)) * QKV_LD + h * D_HEAD + 2 * p;

    float x0 = (float)qkv[base], x1 = (float)qkv[base + 1];
    qkv[base]     = (__bf16)(c * x0 - sn * x1);
    qkv[base + 1] = (__bf16)(sn * x0 + c * x1);

    size_t kb = base + D_MODEL;
    x0 = (float)qkv[kb]; x1 = (float)qkv[kb + 1];
    qkv[kb]     = (__bf16)(c * x0 - sn * x1);
    qkv[kb + 1] = (__bf16)(sn * x0 + c * x1);
}

// ---------------------------------------------------------------------------
// Causal flash attention. One wave per (b, h, 32-query tile).
// ---------------------------------------------------------------------------
__global__ __launch_bounds__(32)
void flash_attn(const __bf16* __restrict__ qkv, __bf16* __restrict__ ctx, int S) {
    __shared__ __bf16 P[32][48];        // 32x32 tile, padded rows (96B)

    int lane = threadIdx.x;
    int qt = blockIdx.x, h = blockIdx.y, b = blockIdx.z;
    int q0 = qt * 32;
    size_t rowbase = (size_t)b * S;
    const __bf16* Q  = qkv + rowbase * QKV_LD + h * D_HEAD;
    const __bf16* Kp = Q + D_MODEL;
    const __bf16* Vp = Q + 2 * D_MODEL;
    const float scale = 0.125f;         // 1/sqrt(64)

    bf16x16 aq[2][2];
#pragma unroll
    for (int mh = 0; mh < 2; ++mh)
#pragma unroll
        for (int ks = 0; ks < 2; ++ks)
            aq[mh][ks] = load_a_frag(Q + (size_t)(q0 + mh * 16) * QKV_LD + ks * 32, QKV_LD, lane);

    f32x8 Oacc[2][4] = {};
    float mrow[2][8], lrow[2][8];
#pragma unroll
    for (int mh = 0; mh < 2; ++mh)
#pragma unroll
        for (int r = 0; r < 8; ++r) { mrow[mh][r] = -1e30f; lrow[mh][r] = 0.0f; }

    int half = lane >> 4, ln = lane & 15;

    for (int kt = 0; kt <= qt; ++kt) {
        int k0 = kt * 32;
        __builtin_prefetch(Vp + (size_t)(k0 + lane) * QKV_LD, 0, 1);

        // ---- scores = Q K^T ----
        bf16x16 bk[2][2];
#pragma unroll
        for (int nh = 0; nh < 2; ++nh)
#pragma unroll
            for (int ks = 0; ks < 2; ++ks)
                bk[nh][ks] = load_b_frag(Kp + (size_t)(k0 + nh * 16) * QKV_LD + ks * 32, QKV_LD, lane);

        f32x8 sc[2][2];
#pragma unroll
        for (int mh = 0; mh < 2; ++mh)
#pragma unroll
            for (int nh = 0; nh < 2; ++nh) {
                f32x8 z = {};
                z = wmma_bf16(aq[mh][0], bk[nh][0], z);
                z = wmma_bf16(aq[mh][1], bk[nh][1], z);
                sc[mh][nh] = z;
            }

        // ---- mask + online softmax ----
#pragma unroll
        for (int mh = 0; mh < 2; ++mh) {
#pragma unroll
            for (int r = 0; r < 8; ++r) {
                int row = q0 + mh * 16 + r + 8 * half;
                float v0 = sc[mh][0][r] * scale;
                float v1 = sc[mh][1][r] * scale;
                if (k0 + ln > row)      v0 = -1e30f;
                if (k0 + 16 + ln > row) v1 = -1e30f;
                float mx = fmaxf(v0, v1);
#pragma unroll
                for (int d = 8; d >= 1; d >>= 1)
                    mx = fmaxf(mx, __shfl_xor(mx, d, 32));
                float mnew  = fmaxf(mrow[mh][r], mx);
                float alpha = __expf(mrow[mh][r] - mnew);
                float p0 = __expf(v0 - mnew);
                float p1 = __expf(v1 - mnew);
                float ps = p0 + p1;
#pragma unroll
                for (int d = 8; d >= 1; d >>= 1)
                    ps += __shfl_xor(ps, d, 32);
                lrow[mh][r] = lrow[mh][r] * alpha + ps;
                mrow[mh][r] = mnew;
#pragma unroll
                for (int g = 0; g < 4; ++g) Oacc[mh][g][r] *= alpha;
                P[mh * 16 + r + 8 * half][ln]      = (__bf16)p0;
                P[mh * 16 + r + 8 * half][16 + ln] = (__bf16)p1;
            }
        }
        __syncthreads();

        // ---- O += P V ----
        bf16x16 ap[2];
#pragma unroll
        for (int mh = 0; mh < 2; ++mh)
            ap[mh] = load_a_frag(&P[mh * 16][0], 48, lane);

        int kb = (lane >> 4) * 16;
#pragma unroll
        for (int g = 0; g < 4; ++g) {
            const __bf16* vp = Vp + (size_t)(k0 + kb) * QKV_LD + g * 16 + ln;
            bf16x16 bv;
#pragma unroll
            for (int e = 0; e < 16; ++e) bv[e] = vp[(size_t)e * QKV_LD];
#pragma unroll
            for (int mh = 0; mh < 2; ++mh)
                Oacc[mh][g] = wmma_bf16(ap[mh], bv, Oacc[mh][g]);
        }
        __syncthreads();
    }

    // ---- finalize: ctx[b, s, h*64 + d] ----
#pragma unroll
    for (int mh = 0; mh < 2; ++mh)
#pragma unroll
        for (int r = 0; r < 8; ++r) {
            float inv = 1.0f / lrow[mh][r];
            size_t row = rowbase + q0 + mh * 16 + r + 8 * half;
#pragma unroll
            for (int g = 0; g < 4; ++g)
                ctx[row * D_MODEL + h * D_HEAD + g * 16 + ln] =
                    (__bf16)(Oacc[mh][g][r] * inv);
        }
}

// ---------------------------------------------------------------------------
// Launcher
// ---------------------------------------------------------------------------
extern "C" void kernel_launch(void* const* d_in, const int* in_sizes, int n_in,
                              void* d_out, int out_size, void* d_ws, size_t ws_size,
                              hipStream_t stream) {
    const float* x    = (const float*)d_in[0];   // [B,S,1024] fp32
    const int*   pos  = (const int*)  d_in[1];   // [S] int32
    const float* wqkv = (const float*)d_in[2];   // [3072,1024] fp32
    const float* wout = (const float*)d_in[3];   // [1024,1024] fp32

    const int S = in_sizes[1];                   // 2048
    const int B = in_sizes[0] / (S * D_MODEL);   // 4
    const int M = B * S;                         // 8192

    char* ws = (char*)d_ws;
    __bf16* Xbf    = (__bf16*)(ws);                                  // 16 MB
    __bf16* Wqkvbf = (__bf16*)(ws + (size_t)16 * 1024 * 1024);       //  6 MB
    __bf16* Woutbf = (__bf16*)(ws + (size_t)22 * 1024 * 1024);       //  2 MB
    __bf16* QKVbf  = (__bf16*)(ws + (size_t)24 * 1024 * 1024);       // 48 MB
    __bf16* CTXbf  = (__bf16*)(ws + (size_t)72 * 1024 * 1024);       // 16 MB

    int nx = M * D_MODEL, nq = 3 * D_MODEL * D_MODEL, no = D_MODEL * D_MODEL;
    cvt_f32_bf16<<<(nx + 255) / 256, 256, 0, stream>>>(x,    Xbf,    nx);
    cvt_f32_bf16<<<(nq + 255) / 256, 256, 0, stream>>>(wqkv, Wqkvbf, nq);
    cvt_f32_bf16<<<(no + 255) / 256, 256, 0, stream>>>(wout, Woutbf, no);

    // QKV projection: [M,1024] x [3072,1024]^T -> bf16 [M,3072]
    gemm_nt<true><<<dim3(QKV_LD / 64, M / 128), 256, 0, stream>>>(
        Xbf, Wqkvbf, QKVbf, M, QKV_LD, D_MODEL);

    // RoPE on Q,K in place
    int nr = M * NUM_HEADS * (D_HEAD / 2);
    rope_qk<<<nr / 256, 256, 0, stream>>>(QKVbf, pos, S);

    // Causal flash attention -> merged ctx bf16 [M,1024]
    flash_attn<<<dim3(S / 32, NUM_HEADS, B), 32, 0, stream>>>(QKVbf, CTXbf, S);

    // Output projection: [M,1024] x [1024,1024]^T -> fp32 d_out
    gemm_nt<false><<<dim3(D_MODEL / 64, M / 128), 256, 0, stream>>>(
        CTXbf, Woutbf, d_out, M, D_MODEL, D_MODEL);
}
